// GiG_30416958390763
// MI455X (gfx1250) — matrix operand order repaired and checked by reading, not compiled, not run on hardware
//
#include <hip/hip_runtime.h>
#include <hip/hip_bf16.h>

// ---------------- CDNA5 WMMA types ----------------
typedef __attribute__((ext_vector_type(16))) _Float16 v16h;
typedef __attribute__((ext_vector_type(8)))  float    v8f;

#define N_NODES 65536
#define N_EDGES 262144
#define P_GRAPH 1024

__device__ __forceinline__ float gelu_f(float x) {
  return 0.5f * x * (1.0f + erff(x * 0.70710678118654752440f));
}

// =====================================================================
// Generic GEMM: C[M,N] = act( A[M,K] @ opB(B) + bias )
// f32 in/out, f16 WMMA compute (v_wmma_f32_16x16x32_f16), f32 accumulate.
// Block = 128 threads (4 waves) -> 64(M) x 32(N) C tile; each wave does a
// 16x32 strip (A fragment reused across two WMMAs). B tile staged in LDS
// (f16, stride 40 halves: 16B-aligned ds_load_b128 reads, conflict-free
// banks (20*mr)%64). Main K loop is branch-free (no per-lane guards, no
// uniform tail checks); K%32 tail handled once after the loop.
// TRANSB=0: B is [K,N] row-major.  TRANSB=1: B is [N,K] row-major (B^T).
// ACT: 0 = none, 1 = relu.
// =====================================================================
#define BS_STRIDE 40

template<int TRANSB, int ACT>
__global__ __launch_bounds__(128)
void gemm_wmma_k(const float* __restrict__ A, const float* __restrict__ B,
                 const float* __restrict__ bias, float* __restrict__ C,
                 int M, int N, int K)
{
  __shared__ _Float16 Bs[32 * BS_STRIDE];   // [nn][kk] at nn*BS_STRIDE + kk
  const int tid  = threadIdx.x;
  const int lane = tid & 31;
  const int wave = tid >> 5;
  const int m0 = (blockIdx.y * 4 + wave) * 16;
  const int n0 = blockIdx.x * 32;
  const int mr = lane & 15;
  const int hi = lane >> 4;                 // half-wave select
  const int arow = (m0 + mr < M) ? (m0 + mr) : (M - 1);   // clamp (loads only; stores guarded)
  const float* __restrict__ Arow = A + (size_t)arow * K;
  const bool nedge = (n0 + 32 > N);         // block-uniform (only logits GEMM, N=8)
  const int Kfull = K & ~31;
  v8f acc0 = {}, acc1 = {};

  // ---------------- main loop: full 32-K tiles, branch-free ----------------
  for (int k0 = 0; k0 < Kfull; k0 += 32) {
    __syncthreads();
    if (!nedge) {
      if (TRANSB) {
        // B[N,K]: float2 along kk (coalesced); adjacent halves -> merged ds_store_b32
#pragma unroll
        for (int t = tid; t < 512; t += 128) {
          int kk2 = (t & 15) << 1, nn = t >> 4;
          float2 bv = *(const float2*)&B[(size_t)(n0 + nn) * K + (k0 + kk2)];
          _Float16* d = &Bs[nn * BS_STRIDE + kk2];
          d[0] = (_Float16)bv.x;
          d[1] = (_Float16)bv.y;
        }
      } else {
        // B[K,N]: float4 along nn (global_load_b128); 4x ds_store_b16 fan-out
#pragma unroll
        for (int t = tid; t < 256; t += 128) {
          int nn4 = (t & 7) << 2, kk = t >> 3;
          float4 bv = *(const float4*)&B[(size_t)(k0 + kk) * N + (n0 + nn4)];
          _Float16* d = &Bs[nn4 * BS_STRIDE + kk];
          d[0 * BS_STRIDE] = (_Float16)bv.x;
          d[1 * BS_STRIDE] = (_Float16)bv.y;
          d[2 * BS_STRIDE] = (_Float16)bv.z;
          d[3 * BS_STRIDE] = (_Float16)bv.w;
        }
      }
    } else {
      // N-edge tile (cold): scalar guarded staging, zero-padded
#pragma unroll
      for (int t = tid; t < 1024; t += 128) {
        int kk = TRANSB ? (t & 31) : (t >> 5);
        int nn = TRANSB ? (t >> 5) : (t & 31);
        float bv = 0.0f;
        if ((n0 + nn) < N)
          bv = TRANSB ? B[(size_t)(n0 + nn) * K + (k0 + kk)]
                      : B[(size_t)(k0 + kk) * N + (n0 + nn)];
        Bs[nn * BS_STRIDE + kk] = (_Float16)bv;
      }
    }
    __syncthreads();
    // A fragment direct from global (f32 -> f16), contiguous 8-float runs.
    // A layout: lanes 0-15 hold K 0..15, lanes 16-31 hold K 16..31 (matches B)
    v16h af;
    const float* __restrict__ a0 = Arow + k0 + 16 * hi;
#pragma unroll
    for (int i = 0; i < 8; ++i) af[i] = (_Float16)a0[i];
#pragma unroll
    for (int i = 0; i < 8; ++i) af[8 + i] = (_Float16)a0[8 + i];
    __builtin_prefetch(Arow + k0 + 128, 0, 0);                    // global_prefetch_b8
    // B fragments from LDS: lane holds column n; K = 16*hi + 0..15 contiguous -> ds_load_b128 x2
    v16h bf0, bf1;
    const _Float16* __restrict__ bp0 = &Bs[mr * BS_STRIDE + 16 * hi];
    const _Float16* __restrict__ bp1 = &Bs[(mr + 16) * BS_STRIDE + 16 * hi];
#pragma unroll
    for (int i = 0; i < 16; ++i) bf0[i] = bp0[i];
#pragma unroll
    for (int i = 0; i < 16; ++i) bf1[i] = bp1[i];
    acc0 = __builtin_amdgcn_wmma_f32_16x16x32_f16(false, af, false, bf0, (short)0, acc0, false, false);
    acc1 = __builtin_amdgcn_wmma_f32_16x16x32_f16(false, af, false, bf1, (short)0, acc1, false, false);
  }

  // ---------------- K tail (runs once; only layer-0 K=129) ----------------
  if (Kfull < K) {
    const int k0 = Kfull;
    __syncthreads();
#pragma unroll
    for (int t = tid; t < 1024; t += 128) {
      int kk = TRANSB ? (t & 31) : (t >> 5);
      int nn = TRANSB ? (t >> 5) : (t & 31);
      float bv = 0.0f;
      if ((n0 + nn) < N && (k0 + kk) < K)
        bv = TRANSB ? B[(size_t)(n0 + nn) * K + (k0 + kk)]
                    : B[(size_t)(k0 + kk) * N + (n0 + nn)];
      Bs[nn * BS_STRIDE + kk] = (_Float16)bv;
    }
    __syncthreads();
    v16h af;
#pragma unroll
    for (int i = 0; i < 16; ++i) {
      int ka = k0 + i + 16 * hi;
      af[i] = (_Float16)((ka < K) ? Arow[ka] : 0.0f);
    }
    v16h bf0, bf1;
    const _Float16* __restrict__ bp0 = &Bs[mr * BS_STRIDE + 16 * hi];
    const _Float16* __restrict__ bp1 = &Bs[(mr + 16) * BS_STRIDE + 16 * hi];
#pragma unroll
    for (int i = 0; i < 16; ++i) bf0[i] = bp0[i];
#pragma unroll
    for (int i = 0; i < 16; ++i) bf1[i] = bp1[i];
    acc0 = __builtin_amdgcn_wmma_f32_16x16x32_f16(false, af, false, bf0, (short)0, acc0, false, false);
    acc1 = __builtin_amdgcn_wmma_f32_16x16x32_f16(false, af, false, bf1, (short)0, acc1, false, false);
  }

  // ---------------- store: acc[r] = C[m0 + r + 8*hi][n] ----------------
  const int cn0 = n0 + mr;
  const int cn1 = n0 + mr + 16;
  const float bv0 = (bias && cn0 < N) ? bias[cn0] : 0.0f;
  const float bv1 = (bias && cn1 < N) ? bias[cn1] : 0.0f;
  if (!nedge) {
#pragma unroll
    for (int r = 0; r < 8; ++r) {
      int cm = m0 + r + 8 * hi;
      if (cm < M) {
        float v0 = acc0[r] + bv0, v1 = acc1[r] + bv1;
        if (ACT == 1) { v0 = fmaxf(v0, 0.0f); v1 = fmaxf(v1, 0.0f); }
        C[(size_t)cm * N + cn0] = v0;
        C[(size_t)cm * N + cn1] = v1;
      }
    }
  } else {
#pragma unroll
    for (int r = 0; r < 8; ++r) {
      int cm = m0 + r + 8 * hi;
      if (cm < M) {
        if (cn0 < N) {
          float v0 = acc0[r] + bv0;
          if (ACT == 1) v0 = fmaxf(v0, 0.0f);
          C[(size_t)cm * N + cn0] = v0;
        }
        if (cn1 < N) {
          float v1 = acc1[r] + bv1;
          if (ACT == 1) v1 = fmaxf(v1, 0.0f);
          C[(size_t)cm * N + cn1] = v1;
        }
      }
    }
  }
}

// ---------------- F1: degree / input assembly ----------------
__global__ void deg_count_k(const int* __restrict__ col, float* __restrict__ deg, int E)
{
  for (int e = blockIdx.x * blockDim.x + threadIdx.x; e < E; e += gridDim.x * blockDim.x)
    unsafeAtomicAdd(&deg[col[e]], 1.0f);
}

__global__ void inv_deg_k(const float* __restrict__ deg, float* __restrict__ inv, int n)
{
  for (int i = blockIdx.x * blockDim.x + threadIdx.x; i < n; i += gridDim.x * blockDim.x)
    inv[i] = rsqrtf(deg[i] + 1.0f);   // self-loop included
}

__global__ void edge_coef_k(const int* __restrict__ row, const int* __restrict__ col,
                            const float* __restrict__ inv, float* __restrict__ coef, int E)
{
  for (int e = blockIdx.x * blockDim.x + threadIdx.x; e < E; e += gridDim.x * blockDim.x)
    coef[e] = inv[row[e]] * inv[col[e]];
}

__global__ void build_h0_k(const float* __restrict__ x, const float* __restrict__ deg,
                           float* __restrict__ h0, int n)
{
  size_t total = (size_t)n * 129;
  for (size_t t = blockIdx.x * (size_t)blockDim.x + threadIdx.x; t < total;
       t += (size_t)gridDim.x * blockDim.x) {
    size_t node = t / 129; int d = (int)(t % 129);
    h0[t] = (d < 128) ? x[node * 128 + d] : deg[node];
  }
}

// out[n] = inv[n]^2 * hW[n] + bias   (self-loop term; initializes scatter target)
__global__ void gcn_combine_k(const float* __restrict__ hW, const float* __restrict__ inv,
                              const float* __restrict__ bias, float* __restrict__ out,
                              int rows, int cols)
{
  int cols4 = cols >> 2;
  size_t total = (size_t)rows * cols4;
  for (size_t t = blockIdx.x * (size_t)blockDim.x + threadIdx.x; t < total;
       t += (size_t)gridDim.x * blockDim.x) {
    size_t n = t / cols4; int j4 = (int)(t % cols4) << 2;
    float iv = inv[n]; float iv2 = iv * iv;
    float4 hv = *(const float4*)&hW[n * cols + j4];
    float4 o;
    o.x = iv2 * hv.x + bias[j4 + 0];
    o.y = iv2 * hv.y + bias[j4 + 1];
    o.z = iv2 * hv.z + bias[j4 + 2];
    o.w = iv2 * hv.w + bias[j4 + 3];
    *(float4*)&out[n * cols + j4] = o;
  }
}

// out[col[e]] += coef[e]*hW[row[e]]   (symmetric-normalized propagate, L2 atomics)
__global__ void gcn_scatter_k(const float* __restrict__ hW, const float* __restrict__ coef,
                              const int* __restrict__ row, const int* __restrict__ col,
                              float* __restrict__ out, int E, int cols)
{
  int cols4 = cols >> 2;
  size_t total = (size_t)E * cols4;
  for (size_t t = blockIdx.x * (size_t)blockDim.x + threadIdx.x; t < total;
       t += (size_t)gridDim.x * blockDim.x) {
    int e = (int)(t / cols4); int j4 = (int)(t % cols4) << 2;
    int r = row[e], c = col[e];
    float cf = coef[e];
    float4 hv = *(const float4*)&hW[(size_t)r * cols + j4];
    float* op = &out[(size_t)c * cols + j4];
    unsafeAtomicAdd(op + 0, cf * hv.x);
    unsafeAtomicAdd(op + 1, cf * hv.y);
    unsafeAtomicAdd(op + 2, cf * hv.z);
    unsafeAtomicAdd(op + 3, cf * hv.w);
  }
}

// ---------------- BatchNorm (batch statistics, biased var) ----------------
__global__ void bn_stats_k(const float* __restrict__ h, float* __restrict__ sum,
                           float* __restrict__ sq, int rows, int cols)
{
  int j = threadIdx.x;
  if (j >= cols) return;
  int chunk = (rows + gridDim.x - 1) / gridDim.x;
  int r0 = blockIdx.x * chunk;
  int r1 = r0 + chunk; if (r1 > rows) r1 = rows;
  float s = 0.0f, s2 = 0.0f;
  for (int r = r0; r < r1; ++r) {           // coalesced: thread j reads column j
    float v = h[(size_t)r * cols + j];
    s += v; s2 += v * v;
  }
  unsafeAtomicAdd(&sum[j], s);
  unsafeAtomicAdd(&sq[j], s2);
}

__global__ void bn_apply_k(float* __restrict__ h, const float* __restrict__ sum,
                           const float* __restrict__ sq, const float* __restrict__ g,
                           const float* __restrict__ b, int rows, int cols, int leaky)
{
  int cols4 = cols >> 2;
  size_t total = (size_t)rows * cols4;
  float invR = 1.0f / (float)rows;
  for (size_t t = blockIdx.x * (size_t)blockDim.x + threadIdx.x; t < total;
       t += (size_t)gridDim.x * blockDim.x) {
    int j4 = (int)(t % cols4) << 2;
    float4 hv = *(const float4*)&h[(t / cols4) * cols + j4];
    float o[4] = {hv.x, hv.y, hv.z, hv.w};
#pragma unroll
    for (int u = 0; u < 4; ++u) {
      int j = j4 + u;
      float m = sum[j] * invR;
      float v = sq[j] * invR - m * m;
      float y = (o[u] - m) * rsqrtf(v + 1e-5f) * g[j] + b[j];
      o[u] = leaky ? (y > 0.0f ? y : 0.2f * y) : fmaxf(y, 0.0f);
    }
    float4 ov = {o[0], o[1], o[2], o[3]};
    *(float4*)&h[(t / cols4) * cols + j4] = ov;
  }
}

// ---------------- pooling ----------------
__global__ void pool_k(const float* __restrict__ emb, const int* __restrict__ batch,
                       float* __restrict__ g, int rows, int cols)
{
  int cols4 = cols >> 2;
  size_t total = (size_t)rows * cols4;
  for (size_t t = blockIdx.x * (size_t)blockDim.x + threadIdx.x; t < total;
       t += (size_t)gridDim.x * blockDim.x) {
    size_t n = t / cols4; int j4 = (int)(t % cols4) << 2;
    float4 hv = *(const float4*)&emb[n * cols + j4];
    float* op = &g[(size_t)batch[n] * cols + j4];
    unsafeAtomicAdd(op + 0, hv.x);
    unsafeAtomicAdd(op + 1, hv.y);
    unsafeAtomicAdd(op + 2, hv.z);
    unsafeAtomicAdd(op + 3, hv.w);
  }
}

// ---------------- LayerNorm + exact GELU (in place), blockDim == cols ----------------
__global__ void ln_gelu_k(float* __restrict__ x, const float* __restrict__ g,
                          const float* __restrict__ b, int cols)
{
  __shared__ float sm[512];
  const int row = blockIdx.x, j = threadIdx.x;
  float v = x[(size_t)row * cols + j];
  sm[j] = v; __syncthreads();
  for (int s = cols >> 1; s > 0; s >>= 1) { if (j < s) sm[j] += sm[j + s]; __syncthreads(); }
  const float m = sm[0] / cols; __syncthreads();
  float d = v - m;
  sm[j] = d * d; __syncthreads();
  for (int s = cols >> 1; s > 0; s >>= 1) { if (j < s) sm[j] += sm[j + s]; __syncthreads(); }
  const float var = sm[0] / cols;
  float y = d * rsqrtf(var + 1e-5f) * g[j] + b[j];
  x[(size_t)row * cols + j] = gelu_f(y);
}

// ---------------- F2: adjacency ----------------
__global__ void rownorm_k(const float* __restrict__ l, float* __restrict__ rn, int cols)
{
  __shared__ float sm[64];
  int row = blockIdx.x, j = threadIdx.x;
  float v = l[(size_t)row * cols + j];
  sm[j] = v * v; __syncthreads();
  for (int s = cols >> 1; s > 0; s >>= 1) { if (j < s) sm[j] += sm[j + s]; __syncthreads(); }
  if (j == 0) rn[row] = sm[0];
}

// A = sigmoid(-temp*dist + theta), diag forced to 1; Am = A * (A > 0.5)
__global__ void adj_k(const float* __restrict__ G, const float* __restrict__ rn,
                      const float* __restrict__ temp, const float* __restrict__ theta,
                      float* __restrict__ Aout, float* __restrict__ Am, int P)
{
  size_t total = (size_t)P * P;
  float tp = temp[0], th = theta[0];
  for (size_t t = blockIdx.x * (size_t)blockDim.x + threadIdx.x; t < total;
       t += (size_t)gridDim.x * blockDim.x) {
    int i = (int)(t / P), j = (int)(t % P);
    float a;
    if (i == j) a = 1.0f;
    else {
      float d2 = fmaxf(rn[i] + rn[j] - 2.0f * G[t], 0.0f);    // == sum((li-lj)^2)
      float dist = sqrtf(d2 + 1e-6f);
      a = 1.0f / (1.0f + expf(tp * dist - th));               // sigmoid(-tp*dist+th)
    }
    Aout[t] = a;
    Am[t] = (a > 0.5f) ? a : 0.0f;
  }
}

__global__ void row_sum_k(const float* __restrict__ Am, float* __restrict__ out, int P)
{
  __shared__ float sm[256];
  int i = blockIdx.x;
  float s = 0.0f;
  for (int j = threadIdx.x; j < P; j += 256) s += Am[(size_t)i * P + j];
  sm[threadIdx.x] = s; __syncthreads();
  for (int st = 128; st > 0; st >>= 1) { if (threadIdx.x < st) sm[threadIdx.x] += sm[threadIdx.x + st]; __syncthreads(); }
  if (threadIdx.x == 0) out[i] = sm[0];
}

// q_raw[j] = sum_i exp(-(d_bar[i] - j)^2)
__global__ void q_k(const float* __restrict__ dbar, float* __restrict__ qraw, int P)
{
  __shared__ float sm[256];
  int j = blockIdx.x;
  float s = 0.0f;
  for (int i = threadIdx.x; i < P; i += 256) {
    float d = dbar[i] - (float)j;
    s += expf(-d * d);
  }
  sm[threadIdx.x] = s; __syncthreads();
  for (int st = 128; st > 0; st >>= 1) { if (threadIdx.x < st) sm[threadIdx.x] += sm[threadIdx.x + st]; __syncthreads(); }
  if (threadIdx.x == 0) qraw[j] = sm[0];
}

__global__ void kl_k(const float* __restrict__ qraw, const float* __restrict__ mu,
                     const float* __restrict__ sigma, float* __restrict__ out)
{
  __shared__ float sm[256];
  __shared__ float qsum_s, rsum_s;
  int t = threadIdx.x;
  float mu_ = mu[0], sg = sigma[0];
  float inv2s2 = 1.0f / (2.0f * sg * sg);
  float qs = 0.0f, rs = 0.0f;
  for (int j = t; j < P_GRAPH; j += 256) {
    qs += qraw[j];
    float d = (float)j - mu_;
    rs += expf(-d * d * inv2s2);
  }
  sm[t] = qs; __syncthreads();
  for (int s = 128; s > 0; s >>= 1) { if (t < s) sm[t] += sm[t + s]; __syncthreads(); }
  if (t == 0) qsum_s = sm[0];
  __syncthreads();
  sm[t] = rs; __syncthreads();
  for (int s = 128; s > 0; s >>= 1) { if (t < s) sm[t] += sm[t + s]; __syncthreads(); }
  if (t == 0) rsum_s = sm[0];
  __syncthreads();
  float Q = qsum_s + 1e-8f, R = rsum_s + 1e-8f;
  float kl = 0.0f;
  for (int j = t; j < P_GRAPH; j += 256) {
    float q = qraw[j] / Q;
    float d = (float)j - mu_;
    float r = expf(-d * d * inv2s2) / R;
    kl += q * logf(q / (r + 1e-8f) + 1e-8f);
  }
  sm[t] = kl; __syncthreads();
  for (int s = 128; s > 0; s >>= 1) { if (t < s) sm[t] += sm[t + s]; __syncthreads(); }
  if (t == 0) out[0] = fminf(fmaxf(sm[0], 0.0f), 10.0f);
}

__global__ void invp_k(const float* __restrict__ degp, float* __restrict__ invp, int P)
{
  for (int i = blockIdx.x * blockDim.x + threadIdx.x; i < P; i += gridDim.x * blockDim.x)
    invp[i] = rsqrtf(degp[i]);                 // degp > 0 (diag self-loop = 1)
}

__global__ void ahat_k(const float* __restrict__ Am, const float* __restrict__ invp,
                       float* __restrict__ Ahat, int P)
{
  size_t total = (size_t)P * P;
  for (size_t t = blockIdx.x * (size_t)blockDim.x + threadIdx.x; t < total;
       t += (size_t)gridDim.x * blockDim.x) {
    int i = (int)(t / P), j = (int)(t % P);
    Ahat[t] = invp[i] * Am[t] * invp[j];
  }
}

// =====================================================================
// host side
// =====================================================================
static inline unsigned gblk(size_t total, unsigned cap = 8192) {
  size_t b = (total + 255) / 256;
  if (b > cap) b = cap;
  if (b == 0) b = 1;
  return (unsigned)b;
}

static void gemm(hipStream_t s, const float* A, const float* B, const float* bias,
                 float* C, int M, int N, int K, int transB, int act)
{
  dim3 grid((N + 31) / 32, (M + 63) / 64);
  dim3 block(128);
  if (transB)       gemm_wmma_k<1, 0><<<grid, block, 0, s>>>(A, B, bias, C, M, N, K);
  else if (act)     gemm_wmma_k<0, 1><<<grid, block, 0, s>>>(A, B, bias, C, M, N, K);
  else              gemm_wmma_k<0, 0><<<grid, block, 0, s>>>(A, B, bias, C, M, N, K);
}

extern "C" void kernel_launch(void* const* d_in, const int* in_sizes, int n_in,
                              void* d_out, int out_size, void* d_ws, size_t ws_size,
                              hipStream_t stream)
{
  const int N = N_NODES, E = N_EDGES, P = P_GRAPH;
  // ---- inputs (setup_inputs dict order, params dict insertion order) ----
  const float* x     = (const float*)d_in[0];
  const int*   ei    = (const int*)d_in[1];
  const int*   rowi  = ei;
  const int*   coli  = ei + E;
  const int*   batch = (const int*)d_in[2];
  const float *f1W[4], *f1b[4], *f1g[4], *f1bt[4];
  for (int i = 0; i < 4; ++i) {
    f1W[i]  = (const float*)d_in[3 + 4 * i];
    f1b[i]  = (const float*)d_in[4 + 4 * i];
    f1g[i]  = (const float*)d_in[5 + 4 * i];
    f1bt[i] = (const float*)d_in[6 + 4 * i];
  }
  const float* f2W1   = (const float*)d_in[19];
  const float* f2b1   = (const float*)d_in[20];
  const float* f2lng  = (const float*)d_in[21];
  const float* f2lnb  = (const float*)d_in[22];
  const float* f2W2   = (const float*)d_in[23];
  const float* f2b2   = (const float*)d_in[24];
  const float* temp   = (const float*)d_in[25];
  const float* theta  = (const float*)d_in[26];
  const float* mu     = (const float*)d_in[27];
  const float* sigma  = (const float*)d_in[28];
  const float* f3inW  = (const float*)d_in[29];
  const float* f3inb  = (const float*)d_in[30];
  const float* f3inlng= (const float*)d_in[31];
  const float* f3inlnb= (const float*)d_in[32];
  const float* f3W[2] = {(const float*)d_in[33], (const float*)d_in[35]};
  const float* f3b[2] = {(const float*)d_in[34], (const float*)d_in[36]};
  const float* clfW0  = (const float*)d_in[37];
  const float* clfb0  = (const float*)d_in[38];
  const float* clflng0= (const float*)d_in[39];
  const float* clflnb0= (const float*)d_in[40];
  const float* clfW1  = (const float*)d_in[41];
  const float* clfb1  = (const float*)d_in[42];
  const float* clflng1= (const float*)d_in[43];
  const float* clflnb1= (const float*)d_in[44];
  const float* clfW2  = (const float*)d_in[45];
  const float* clfb2  = (const float*)d_in[46];

  // ---- outputs: logits[P*8] | A[P*P] | kl[1] ----
  float* outF   = (float*)d_out;
  float* logits = outF;
  float* Aout   = outF + (size_t)P * 8;
  float* klout  = outF + (size_t)P * 8 + (size_t)P * P;

  // ---- workspace layout (floats) ----
  float* w = (float*)d_ws;
  size_t o = 0;
  float* deg  = w + o; o += N;
  float* inv  = w + o; o += N;
  float* ecoef= w + o; o += E;
  float* bufA = w + o; o += (size_t)N * 256;   // h (ping) ; holds h0 [N,129] initially
  float* bufW = w + o; o += (size_t)N * 256;   // hW (pong)
  float* gbuf = w + o; o += (size_t)P * 128;
  float* l1   = w + o; o += (size_t)P * 64;
  float* lv   = w + o; o += (size_t)P * 64;
  float* rn   = w + o; o += P;
  float* G    = w + o; o += (size_t)P * P;
  float* Am   = w + o; o += (size_t)P * P;
  float* dbar = w + o; o += P;                  // == degp (Am symmetric)
  float* invp = w + o; o += P;
  float* qraw = w + o; o += P;
  float* Ahat = w + o; o += (size_t)P * P;
  float* hp   = w + o; o += (size_t)P * 256;
  float* hpW  = w + o; o += (size_t)P * 256;
  float* hp2  = w + o; o += (size_t)P * 256;
  float* z    = w + o; o += (size_t)P * 512;
  float* z2   = w + o; o += (size_t)P * 512;
  float* bsum = w + o; o += 512;
  float* bsq  = w + o; o += 512;
  (void)ws_size; (void)in_sizes; (void)n_in; (void)out_size;

  // ================= F1: node-level GCN stack =================
  hipMemsetAsync(deg, 0, (size_t)N * 4, stream);
  deg_count_k<<<gblk(E), 256, 0, stream>>>(coli, deg, E);
  inv_deg_k<<<gblk(N), 256, 0, stream>>>(deg, inv, N);
  edge_coef_k<<<gblk(E), 256, 0, stream>>>(rowi, coli, inv, ecoef, E);
  build_h0_k<<<gblk((size_t)N * 129), 256, 0, stream>>>(x, deg, bufA, N);

  const int dimsK[4] = {129, 256, 256, 256};
  const int dimsN[4] = {256, 256, 256, 128};
  for (int i = 0; i < 4; ++i) {
    int Ki = dimsK[i], Ni = dimsN[i];
    gemm(stream, bufA, f1W[i], nullptr, bufW, N, Ni, Ki, 0, 0);     // hW = h @ W  (WMMA)
    size_t tot4 = (size_t)N * (Ni >> 2);
    gcn_combine_k<<<gblk(tot4), 256, 0, stream>>>(bufW, inv, f1b[i], bufA, N, Ni);
    gcn_scatter_k<<<gblk((size_t)E * (Ni >> 2), 16384), 256, 0, stream>>>(bufW, ecoef, rowi, coli, bufA, E, Ni);
    hipMemsetAsync(bsum, 0, 512 * 4, stream);
    hipMemsetAsync(bsq,  0, 512 * 4, stream);
    bn_stats_k<<<256, Ni, 0, stream>>>(bufA, bsum, bsq, N, Ni);
    bn_apply_k<<<gblk(tot4), 256, 0, stream>>>(bufA, bsum, bsq, f1g[i], f1bt[i], N, Ni,
                                               (i == 3) ? 1 : 0);   // leaky(0.2) at last layer
  }

  // global_add_pool -> g [P,128]
  hipMemsetAsync(gbuf, 0, (size_t)P * 128 * 4, stream);
  pool_k<<<gblk((size_t)N * 32), 256, 0, stream>>>(bufA, batch, gbuf, N, 128);

  // ================= F2: latent population graph =================
  gemm(stream, gbuf, f2W1, f2b1, l1, P, 64, 128, 0, 0);
  ln_gelu_k<<<P, 64, 0, stream>>>(l1, f2lng, f2lnb, 64);
  gemm(stream, l1, f2W2, f2b2, lv, P, 64, 64, 0, 0);
  rownorm_k<<<P, 64, 0, stream>>>(lv, rn, 64);
  gemm(stream, lv, lv, nullptr, G, P, P, 64, 1, 0);                 // Gram = l @ l^T (WMMA, transB)
  adj_k<<<gblk((size_t)P * P), 256, 0, stream>>>(G, rn, temp, theta, Aout, Am, P);
  row_sum_k<<<P, 256, 0, stream>>>(Am, dbar, P);                    // d_bar (= degp, symmetric)
  q_k<<<P, 256, 0, stream>>>(dbar, qraw, P);
  kl_k<<<1, 256, 0, stream>>>(qraw, mu, sigma, klout);

  // ================= F3: classifier on learned graph =================
  gemm(stream, gbuf, f3inW, f3inb, hp, P, 256, 128, 0, 0);
  ln_gelu_k<<<P, 256, 0, stream>>>(hp, f3inlng, f3inlnb, 256);
  invp_k<<<gblk(P), 256, 0, stream>>>(dbar, invp, P);
  ahat_k<<<gblk((size_t)P * P), 256, 0, stream>>>(Am, invp, Ahat, P);
  float* hin = hp; float* hout = hp2;
  for (int i = 0; i < 2; ++i) {
    gemm(stream, hin, f3W[i], nullptr, hpW, P, 256, 256, 0, 0);     // hp @ W
    gemm(stream, Ahat, hpW, f3b[i], hout, P, 256, P, 0, 1);         // relu(Ahat^T @ . + b); Ahat symmetric
    float* t = hin; hin = hout; hout = t;
  }
  gemm(stream, hin, clfW0, clfb0, z, P, 512, 256, 0, 0);
  ln_gelu_k<<<P, 512, 0, stream>>>(z, clflng0, clflnb0, 512);
  gemm(stream, z, clfW1, clfb1, z2, P, 512, 512, 0, 0);
  ln_gelu_k<<<P, 512, 0, stream>>>(z2, clflng1, clflnb1, 512);
  gemm(stream, z2, clfW2, clfb2, logits, P, 8, 512, 0, 0);          // [P, 8]
}
